// RTdetector_4870492914262
// MI455X (gfx1250) — matrix operand order repaired and checked by reading, not compiled, not run on hardware
//
#include <hip/hip_runtime.h>

// ---------------------------------------------------------------------------
// MI455X (gfx1250) implementation. All dense projections use
// v_wmma_f32_16x16x32_f16 (wave32, 16x16 D tile per wave, f32 accum).
// Activations stored f16 for WMMA inputs, f32 for residual/LN precision.
// ---------------------------------------------------------------------------

typedef __attribute__((ext_vector_type(16))) _Float16 v16h;
typedef __attribute__((ext_vector_type(8)))  _Float16 h8;
typedef __attribute__((ext_vector_type(8)))  float    v8f;

union AV { v16h v; h8 h[2]; };

#define EPI_F32    0
#define EPI_F16    1
#define EPI_RELU16 2   // relu, write f16 with zero-pad to ld16 (=32) columns
#define EPI_SIG    3   // sigmoid, write f32

// Y[N,OUT] = X[N,K] @ Wt[OUT,K]^T + bias.  K multiple of 32, N/OUT mult of 16.
// One wave computes one 16x16 tile of Y.
// WMMA operand layouts per CDNA5 ISA 7.12.2:
//   A 16x32 f16 : lane l -> row (l&15); halves 0..7 = K [8g..8g+7],
//                 halves 8..15 = K [16+8g..16+8g+7], g = l>>4
//   B 32x16 f16 : lane l -> col (l&15); same K order as A
//   C/D 16x16 f32: vgpr r -> row (r + 8g), col (l&15)
__global__ void gemm_wmma_kernel(const _Float16* __restrict__ X,
                                 const _Float16* __restrict__ Wt,
                                 const float*    __restrict__ bias,
                                 float*          __restrict__ Y32,
                                 _Float16*       __restrict__ Y16,
                                 int N, int K, int OUT, int ld16, int epi)
{
  const int lane   = threadIdx.x & 31;
  const int wv     = threadIdx.x >> 5;
  const int tilesN = OUT >> 4;
  const int tile   = blockIdx.x * (blockDim.x >> 5) + wv;
  const int tm     = tile / tilesN;
  const int tn     = tile - tm * tilesN;
  if (tm * 16 >= N) return;                 // wave-uniform exit (EXEC stays full)
  const int m = lane & 15;
  const int g = lane >> 4;
  const _Float16* ar = X  + (size_t)(tm * 16 + m) * K;
  const _Float16* br = Wt + (size_t)(tn * 16 + m) * K;
  v8f acc = {};
  for (int k0 = 0; k0 < K; k0 += 32) {
    AV a, b;
    a.h[0] = *(const h8*)(ar + k0 + 8 * g);
    a.h[1] = *(const h8*)(ar + k0 + 16 + 8 * g);
    b.h[0] = *(const h8*)(br + k0 + 8 * g);
    b.h[1] = *(const h8*)(br + k0 + 16 + 8 * g);
    acc = __builtin_amdgcn_wmma_f32_16x16x32_f16(false, a.v, false, b.v,
                                                 (short)0, acc, false, false);
  }
  const int   n  = tn * 16 + m;
  const float bn = bias ? bias[n] : 0.0f;
#pragma unroll
  for (int r = 0; r < 8; ++r) {
    const int mm  = tm * 16 + r + 8 * g;
    float     val = acc[r] + bn;
    if (epi == EPI_F32) {
      Y32[(size_t)mm * OUT + n] = val;
    } else if (epi == EPI_F16) {
      Y16[(size_t)mm * OUT + n] = (_Float16)val;
    } else if (epi == EPI_RELU16) {          // OUT==16 here, tn==0
      float rv = val > 0.0f ? val : 0.0f;
      Y16[(size_t)mm * ld16 + m]      = (_Float16)rv;
      Y16[(size_t)mm * ld16 + 16 + m] = (_Float16)0.0f;
    } else {                                  // EPI_SIG
      Y32[(size_t)mm * OUT + n] = 1.0f / (1.0f + __expf(-val));
    }
  }
}

// Weight f32 -> f16, row-major [rows,cols] -> [rows,pcols] (zero pad).
__global__ void pack_w16_kernel(const float* __restrict__ src,
                                _Float16* __restrict__ dst,
                                int rows, int cols, int pcols)
{
  int i = blockIdx.x * blockDim.x + threadIdx.x;
  if (i >= rows * pcols) return;
  int r = i / pcols, c = i - r * pcols;
  dst[i] = (_Float16)(c < cols ? src[r * cols + c] : 0.0f);
}

__device__ __forceinline__ float pe_val(int w, int d)
{
  // div = exp(-(d&~1) * ln(10000)/128); even d -> sin, odd d -> cos
  float div = __expf(-(float)(d & ~1) * (9.210340372f / 128.0f));
  float ang = (float)w * div;
  return (d & 1) ? __cosf(ang) : __sinf(ang);
}

// Fused: x = concat(src, c) -> per-batch W-dim normalization -> xb,
// s = xb + PE (f32 + f16 out), mean/std out, and both projectors
// (circular conv over L=128, 256->16 relu, 16->{1,10}) -> tau, delta.
// One wave per batch element; 8 waves / 256-thread block.
__global__ void encode_prep_kernel(const float* __restrict__ src,
                                   const float* __restrict__ O1, // null -> c=0
                                   const float* cw_t, const float* l1w_t,
                                   const float* l1b_t, const float* l2w_t,
                                   const float* cw_d, const float* l1w_d,
                                   const float* l1b_d, const float* l2w_d,
                                   float* __restrict__ s32,
                                   _Float16* __restrict__ s16,
                                   float* __restrict__ meanO,
                                   float* __restrict__ stdO,
                                   float* __restrict__ tauO,
                                   float* __restrict__ deltaO,
                                   int Bc)
{
  __shared__ float lds[8][1552];          // per wave: xb[1280] + h[256] + hid[16]
  const int wv = threadIdx.x >> 5, lane = threadIdx.x & 31;
  const int b  = blockIdx.x * 8 + wv;     // grid sized exactly Bc/8
  float* xb  = &lds[wv][0];
  float* hh  = &lds[wv][1280];
  float* hid = &lds[wv][1536];

#pragma unroll
  for (int j = 0; j < 4; ++j) {
    int d = lane * 4 + j;
    float xs[10];
#pragma unroll
    for (int w = 0; w < 10; ++w) {
      float val;
      if (d < 64) {
        val = src[(size_t)w * Bc * 64 + (size_t)b * 64 + d];
      } else {
        float sv = src[(size_t)w * Bc * 64 + (size_t)b * 64 + (d - 64)];
        if (O1) { float e = O1[(size_t)b * 64 + (d - 64)] - sv; val = e * e; }
        else     val = 0.0f;
      }
      xs[w] = val;
    }
    float mu = 0.0f;
#pragma unroll
    for (int w = 0; w < 10; ++w) mu += xs[w];
    mu *= 0.1f;
    float var = 0.0f;
#pragma unroll
    for (int w = 0; w < 10; ++w) { float t = xs[w] - mu; var += t * t; }
    var *= 0.1f;
    float sd  = sqrtf(var + 1e-5f);
    float inv = 1.0f / sd;
    meanO[(size_t)b * 128 + d] = mu;
    stdO [(size_t)b * 128 + d] = sd;
#pragma unroll
    for (int w = 0; w < 10; ++w) {
      float xn = (xs[w] - mu) * inv;
      xb[w * 128 + d] = xn;
      float sv = xn + pe_val(w, d);
      size_t o = (size_t)w * Bc * 128 + (size_t)b * 128 + d;
      s32[o] = sv;
      s16[o] = (_Float16)sv;
    }
  }
  __syncthreads();

  for (int p = 0; p < 2; ++p) {
    const float* cw    = p ? cw_d  : cw_t;
    const float* l1w   = p ? l1w_d : l1w_t;
    const float* l1b   = p ? l1b_d : l1b_t;
    const float* l2w   = p ? l2w_d : l2w_t;
    const float* stats = p ? meanO : stdO;   // tau uses std, delta uses mean
#pragma unroll
    for (int j = 0; j < 4; ++j) {
      int l = lane * 4 + j;
      float c = 0.0f;
#pragma unroll
      for (int w = 0; w < 10; ++w)
#pragma unroll
        for (int kk = 0; kk < 3; ++kk)
          c += xb[w * 128 + ((l + 127 + kk) & 127)] * cw[w * 3 + kk];
      hh[l]       = c;
      hh[128 + l] = stats[(size_t)b * 128 + l];
    }
    __syncthreads();
    if (lane < 16) {
      float acc = l1b[lane];
      for (int i = 0; i < 256; ++i) acc += l1w[lane * 256 + i] * hh[i];
      hid[lane] = acc > 0.0f ? acc : 0.0f;
    }
    __syncthreads();
    if (p == 0) {
      if (lane == 0) {
        float t = 0.0f;
#pragma unroll
        for (int j = 0; j < 16; ++j) t += l2w[j] * hid[j];
        tauO[b] = __expf(t);
      }
    } else {
      if (lane < 10) {
        float t = 0.0f;
#pragma unroll
        for (int j = 0; j < 16; ++j) t += l2w[lane * 16 + j] * hid[j];
        deltaO[(size_t)b * 10 + lane] = t;
      }
    }
    __syncthreads();
  }
}

// Encoder self-attention: head_dim=2, 10 q x 10 k per (b,h). One thread/(b,h).
__global__ void attn_enc_kernel(const _Float16* __restrict__ q,
                                const _Float16* __restrict__ k,
                                const _Float16* __restrict__ v,
                                const float* __restrict__ tau,
                                const float* __restrict__ delta,
                                _Float16* __restrict__ out, int Bc)
{
  int t = blockIdx.x * blockDim.x + threadIdx.x;
  if (t >= Bc * 64) return;
  int b = t >> 6, h = t & 63;
  const size_t base = (size_t)b * 128 + h * 2;
  const size_t ws   = (size_t)Bc * 128;
  float qv[10][2], kv[10][2], vv[10][2];
#pragma unroll
  for (int w = 0; w < 10; ++w) {
    qv[w][0] = (float)q[base + w * ws]; qv[w][1] = (float)q[base + w * ws + 1];
    kv[w][0] = (float)k[base + w * ws]; kv[w][1] = (float)k[base + w * ws + 1];
    vv[w][0] = (float)v[base + w * ws]; vv[w][1] = (float)v[base + w * ws + 1];
  }
  float tb = tau[b] * 0.70710678f;       // tau / sqrt(head_dim=2)
  float dl[10];
#pragma unroll
  for (int j = 0; j < 10; ++j) dl[j] = delta[(size_t)b * 10 + j];
#pragma unroll
  for (int qi = 0; qi < 10; ++qi) {
    float s[10], mx = -1e30f;
#pragma unroll
    for (int j = 0; j < 10; ++j) {
      s[j] = (qv[qi][0] * kv[j][0] + qv[qi][1] * kv[j][1]) * tb + dl[j];
      mx = fmaxf(mx, s[j]);
    }
    float den = 0.0f, o0 = 0.0f, o1 = 0.0f;
#pragma unroll
    for (int j = 0; j < 10; ++j) {
      float a = __expf(s[j] - mx);
      den += a; o0 += a * vv[j][0]; o1 += a * vv[j][1];
    }
    float inv = 1.0f / den;
    out[base + qi * ws]     = (_Float16)(o0 * inv);
    out[base + qi * ws + 1] = (_Float16)(o1 * inv);
  }
}

// Decoder cross-attention: single query per (b,h) against 10 memory keys.
__global__ void attn_cross_kernel(const float* __restrict__ q32,
                                  const _Float16* __restrict__ k2,
                                  const _Float16* __restrict__ v2,
                                  const float* __restrict__ tau,
                                  const float* __restrict__ delta,
                                  _Float16* __restrict__ out, int Bc)
{
  int t = blockIdx.x * blockDim.x + threadIdx.x;
  if (t >= Bc * 64) return;
  int b = t >> 6, h = t & 63;
  const size_t base = (size_t)b * 128 + h * 2;
  const size_t ws   = (size_t)Bc * 128;
  float q0 = q32[base], q1 = q32[base + 1];
  float tb = tau[b] * 0.70710678f;
  float s[10], mx = -1e30f;
#pragma unroll
  for (int w = 0; w < 10; ++w) {
    s[w] = (q0 * (float)k2[base + w * ws] + q1 * (float)k2[base + w * ws + 1]) * tb
         + delta[(size_t)b * 10 + w];
    mx = fmaxf(mx, s[w]);
  }
  float den = 0.0f, o0 = 0.0f, o1 = 0.0f;
#pragma unroll
  for (int w = 0; w < 10; ++w) {
    float a = __expf(s[w] - mx);
    den += a;
    o0 += a * (float)v2[base + w * ws];
    o1 += a * (float)v2[base + w * ws + 1];
  }
  float inv = 1.0f / den;
  out[base]     = (_Float16)(o0 * inv);
  out[base + 1] = (_Float16)(o1 * inv);
}

// res = LN(res + y) over D=128; one wave(32)/token, 4 f32 per lane.
__global__ void add_ln_kernel(const float* __restrict__ y,
                              float* __restrict__ res,
                              const float* __restrict__ gg,
                              const float* __restrict__ bb,
                              _Float16* __restrict__ out16, int Ntok)
{
  const int wv = threadIdx.x >> 5, lane = threadIdx.x & 31;
  const int tok = blockIdx.x * 8 + wv;
  if (tok >= Ntok) return;
  const size_t base = (size_t)tok * 128;
  float x[4], s = 0.0f, sq = 0.0f;
#pragma unroll
  for (int j = 0; j < 4; ++j) {
    int i = lane * 4 + j;
    x[j] = res[base + i] + y[base + i];
    s += x[j]; sq += x[j] * x[j];
  }
#pragma unroll
  for (int off = 16; off > 0; off >>= 1) {
    s  += __shfl_xor(s,  off, 32);
    sq += __shfl_xor(sq, off, 32);
  }
  float mu  = s * (1.0f / 128.0f);
  float var = sq * (1.0f / 128.0f) - mu * mu;
  float inv = rsqrtf(var + 1e-5f);
#pragma unroll
  for (int j = 0; j < 4; ++j) {
    int i = lane * 4 + j;
    float o = (x[j] - mu) * inv * gg[i] + bb[i];
    res[base + i] = o;
    if (out16) out16[base + i] = (_Float16)o;
  }
}

// tgt2 = tile(tgt, 2) along features.
__global__ void tgt_prep_kernel(const float* __restrict__ tgt,
                                float* __restrict__ t32,
                                _Float16* __restrict__ t16, int Bc)
{
  int i = blockIdx.x * blockDim.x + threadIdx.x;
  if (i >= Bc * 128) return;
  int b = i >> 7, d = i & 127;
  float v = tgt[(size_t)b * 64 + (d & 63)];
  t32[i] = v;
  t16[i] = (_Float16)v;
}

// de-normalize: dn = t * std + mean (f16 out for final WMMA GEMM)
__global__ void denorm_kernel(const float* __restrict__ t32,
                              const float* __restrict__ stdB,
                              const float* __restrict__ meanB,
                              _Float16* __restrict__ dn16, int Bc)
{
  int i = blockIdx.x * blockDim.x + threadIdx.x;
  if (i >= Bc * 128) return;
  dn16[i] = (_Float16)(t32[i] * stdB[i] + meanB[i]);
}

// ---------------------------------------------------------------------------
extern "C" void kernel_launch(void* const* d_in, const int* in_sizes, int n_in,
                              void* d_out, int out_size, void* d_ws, size_t ws_size,
                              hipStream_t stream)
{
  (void)in_sizes; (void)n_in; (void)out_size; (void)ws_size;
  const int Bc = 8192;
  const size_t N1 = (size_t)10 * Bc;       // 81920 encoder tokens

  const float* src = (const float*)d_in[0];
  const float* tgt = (const float*)d_in[1];
  // params flattened in jax.tree_util leaf order (alphabetical dict keys):
  // [dec1(26), dec2(26), delta(4), enc(16), fcn_b, fcn_w, tau(4)]
  // dec layout (+o): ca{k_b,k_w,o_b,o_w,q_b,q_w,v_b,v_w}=0..7, f1_b=8,f1_w=9,
  //   f2_b=10,f2_w=11, ln1{b,g}=12,13, ln2{b,g}=14,15, ln3{b,g}=16,17,
  //   sa{k_b,k_w,o_b,o_w,q_b,q_w,v_b,v_w}=18..25
  const float* P[78];
  for (int i = 0; i < 78; ++i) P[i] = (const float*)d_in[2 + i];
  const int ENC = 56;   // enc: f1_b,f1_w,f2_b,f2_w,ln1b,ln1g,ln2b,ln2g,sa(8)

  // ---- workspace carve ----
  char* wsp = (char*)d_ws; size_t off = 0;
  auto alloc = [&](size_t bytes) -> void* {
    void* p = wsp + off; off = (off + bytes + 255) & ~(size_t)255; return p;
  };
  float*    s32   = (float*)   alloc(N1 * 128 * 4);
  _Float16* s16   = (_Float16*)alloc(N1 * 128 * 2);
  _Float16* q16   = (_Float16*)alloc(N1 * 128 * 2);   // enc Q / dec cross-K
  _Float16* k16   = (_Float16*)alloc(N1 * 128 * 2);   // enc K / dec cross-V
  _Float16* v16   = (_Float16*)alloc(N1 * 128 * 2);   // enc V
  _Float16* att16 = (_Float16*)alloc(N1 * 128 * 2);   // attn out / dn16
  float*    y32   = (float*)   alloc(N1 * 128 * 4);   // GEMM f32 out / dec q
  _Float16* hb16  = (_Float16*)alloc(N1 * 32 * 2);    // FFN hidden (K-pad to 32)
  float*    meanB = (float*)   alloc((size_t)Bc * 128 * 4);
  float*    stdB  = (float*)   alloc((size_t)Bc * 128 * 4);
  float*    tauB  = (float*)   alloc((size_t)Bc * 4);
  float*    delB  = (float*)   alloc((size_t)Bc * 10 * 4);
  float*    t32   = (float*)   alloc((size_t)Bc * 128 * 4);
  _Float16* t16   = (_Float16*)alloc((size_t)Bc * 128 * 2);
  auto aw = [&](size_t elems) { return (_Float16*)alloc(elems * 2); };
  _Float16 *encQ = aw(128*128), *encK = aw(128*128), *encV = aw(128*128),
           *encO = aw(128*128), *encF1 = aw(16*128), *encF2 = aw(128*32);
  _Float16 *dW[2][8];   // per dec: saV,saO,caQ,caK,caV,caO,f1,f2
  for (int d = 0; d < 2; ++d) {
    for (int j = 0; j < 6; ++j) dW[d][j] = aw(128*128);
    dW[d][6] = aw(16*128); dW[d][7] = aw(128*32);
  }
  _Float16* fcn16 = aw(64*128);

  auto pack = [&](const float* s, _Float16* d, int r, int c, int pc) {
    int n = r * pc;
    pack_w16_kernel<<<(n + 255) / 256, 256, 0, stream>>>(s, d, r, c, pc);
  };
  auto gemm = [&](const _Float16* X, const _Float16* Wt, const float* bias,
                  float* Y32o, _Float16* Y16o, int N, int K, int OUT,
                  int ld16, int epi) {
    int tiles = (N / 16) * (OUT / 16);
    gemm_wmma_kernel<<<(tiles + 7) / 8, 256, 0, stream>>>(
        X, Wt, bias, Y32o, Y16o, N, K, OUT, ld16, epi);
  };
  auto addln = [&](const float* y, float* res, const float* g, const float* b,
                   _Float16* o16, int Nt) {
    add_ln_kernel<<<(Nt + 7) / 8, 256, 0, stream>>>(y, res, g, b, o16, Nt);
  };

  // ---- pack weights to f16 ----
  pack(P[ENC+13], encQ, 128,128,128);  // enc sa q_w (ENC+12=q_b,+13=q_w)
  pack(P[ENC+9],  encK, 128,128,128);  // enc sa k_w
  pack(P[ENC+15], encV, 128,128,128);  // enc sa v_w
  pack(P[ENC+11], encO, 128,128,128);  // enc sa o_w
  pack(P[ENC+1],  encF1, 16,128,128);  // enc f1_w
  pack(P[ENC+3],  encF2, 128,16,32);   // enc f2_w (pad K 16->32)
  for (int d = 0; d < 2; ++d) {
    int o = d * 26;
    pack(P[o+25], dW[d][0], 128,128,128); // sa v_w
    pack(P[o+21], dW[d][1], 128,128,128); // sa o_w
    pack(P[o+5],  dW[d][2], 128,128,128); // ca q_w
    pack(P[o+1],  dW[d][3], 128,128,128); // ca k_w
    pack(P[o+7],  dW[d][4], 128,128,128); // ca v_w
    pack(P[o+3],  dW[d][5], 128,128,128); // ca o_w
    pack(P[o+9],  dW[d][6], 16,128,128);  // f1_w
    pack(P[o+11], dW[d][7], 128,16,32);   // f2_w padded
  }
  pack(P[73], fcn16, 64, 128, 128);       // fcn_w

  auto run_prep = [&](const float* O1) {
    encode_prep_kernel<<<Bc / 8, 256, 0, stream>>>(
        src, O1,
        P[74], P[76], P[75], P[77],   // tau:   conv_w, l1_w, l1_b, l2_w
        P[52], P[54], P[53], P[55],   // delta: conv_w, l1_w, l1_b, l2_w
        s32, s16, meanB, stdB, tauB, delB, Bc);
  };
  auto run_encoder = [&]() {
    gemm(s16, encQ, P[ENC+12], nullptr, q16, (int)N1, 128, 128, 0, EPI_F16);
    gemm(s16, encK, P[ENC+8],  nullptr, k16, (int)N1, 128, 128, 0, EPI_F16);
    gemm(s16, encV, P[ENC+14], nullptr, v16, (int)N1, 128, 128, 0, EPI_F16);
    attn_enc_kernel<<<(Bc * 64 + 255) / 256, 256, 0, stream>>>(
        q16, k16, v16, tauB, delB, att16, Bc);
    gemm(att16, encO, P[ENC+10], y32, nullptr, (int)N1, 128, 128, 0, EPI_F32);
    addln(y32, s32, P[ENC+5], P[ENC+4], s16, (int)N1);      // ln1 g,b
    gemm(s16, encF1, P[ENC+0], nullptr, hb16, (int)N1, 128, 16, 32, EPI_RELU16);
    gemm(hb16, encF2, P[ENC+2], y32, nullptr, (int)N1, 32, 128, 0, EPI_F32);
    addln(y32, s32, P[ENC+7], P[ENC+6], s16, (int)N1);      // ln2 g,b
    // s32/s16 now hold encoder memory
  };
  auto run_decoder = [&](int d, float* outp) {
    int o = d * 26;
    tgt_prep_kernel<<<(Bc * 128 + 255) / 256, 256, 0, stream>>>(tgt, t32, t16, Bc);
    // self-attn with Lk=1: softmax==1 -> o_proj(v_proj(t))
    gemm(t16, dW[d][0], P[o+24], nullptr, att16, Bc, 128, 128, 0, EPI_F16);
    gemm(att16, dW[d][1], P[o+20], y32, nullptr, Bc, 128, 128, 0, EPI_F32);
    addln(y32, t32, P[o+13], P[o+12], t16, Bc);              // ln1
    // cross-attn
    gemm(t16, dW[d][2], P[o+4], y32, nullptr, Bc, 128, 128, 0, EPI_F32);   // q
    gemm(s16, dW[d][3], P[o+0], nullptr, q16, (int)N1, 128, 128, 0, EPI_F16); // K
    gemm(s16, dW[d][4], P[o+6], nullptr, k16, (int)N1, 128, 128, 0, EPI_F16); // V
    attn_cross_kernel<<<(Bc * 64 + 255) / 256, 256, 0, stream>>>(
        y32, q16, k16, tauB, delB, att16, Bc);
    gemm(att16, dW[d][5], P[o+2], y32, nullptr, Bc, 128, 128, 0, EPI_F32);
    addln(y32, t32, P[o+15], P[o+14], t16, Bc);              // ln2
    // FFN
    gemm(t16, dW[d][6], P[o+8], nullptr, hb16, Bc, 128, 16, 32, EPI_RELU16);
    gemm(hb16, dW[d][7], P[o+10], y32, nullptr, Bc, 32, 128, 0, EPI_F32);
    addln(y32, t32, P[o+17], P[o+16], nullptr, Bc);          // ln3
    // denorm + fcn + sigmoid
    denorm_kernel<<<(Bc * 128 + 255) / 256, 256, 0, stream>>>(
        t32, stdB, meanB, att16, Bc);
    gemm(att16, fcn16, P[72], outp, nullptr, Bc, 128, 64, 0, EPI_SIG);
  };

  float* out = (float*)d_out;
  // branches 1&2 share the same encode (c = 0)
  run_prep(nullptr);
  run_encoder();
  run_decoder(0, out);                       // O1  -> out[0 : B*64)
  run_decoder(1, out + 2 * (size_t)Bc * 64); // O2  -> out[2*B*64 : )
  // branch 3: c_focus = (O1 - src)^2, computed on the fly from d_out
  run_prep(out);
  run_encoder();
  run_decoder(1, out + (size_t)Bc * 64);     // Oh2 -> out[B*64 : 2*B*64)
}
